// KwisatzHaderachBH_25434796327481
// MI455X (gfx1250) — matrix operand order repaired and checked by reading, not compiled, not run on hardware
//
#include <hip/hip_runtime.h>
#include <stdint.h>

// ---------------------------------------------------------------------------
// Problem constants (from the reference)
// ---------------------------------------------------------------------------
#define NPART   8192
#define NBHOLE  64
#define EPEDGE  300000
#define EBEDGE  160000
#define KKTAP   64            // 4*4*4 trilinear taps
#define FOUR_OVER_PI 1.27323954473516268615f

typedef __attribute__((ext_vector_type(16))) __bf16        v16bf;
typedef __attribute__((ext_vector_type(8)))  float         v8f;
typedef __attribute__((ext_vector_type(8)))  unsigned int  v8u;

__device__ __forceinline__ unsigned short f2bf(float x) {
  unsigned u = __float_as_uint(x);
  u += 0x7FFFu + ((u >> 16) & 1u);      // round-to-nearest-even
  return (unsigned short)(u >> 16);
}
__device__ __forceinline__ float signf(float x) {
  return (x > 0.f) ? 1.f : ((x < 0.f) ? -1.f : 0.f);
}

// ---------------------------------------------------------------------------
// Edge geometry: poly6 window + volume-preserving ball->cube + trilinear setup
// ---------------------------------------------------------------------------
struct EdgeGeom { float win; int i0x, i0y, i0z; float fx, fy, fz; };

__device__ __forceinline__ EdgeGeom edge_geom(float dx, float dy, float dz) {
  const float eps = 1e-8f;
  float r2  = dx*dx + dy*dy + dz*dz;
  float w1  = 1.f - r2;
  float win = fminf(fmaxf(w1*w1*w1, 0.f), 1.f);

  float nrm = sqrtf(r2);
  float xy  = dx*dx + dy*dy;
  bool  caps = (1.25f*dz*dz) > xy;
  float s_caps = sqrtf(3.f*nrm / (nrm + fabsf(dz) + eps));
  float s_side = nrm / sqrtf(xy + eps);
  float s = caps ? s_caps : s_side;
  bool  valid = r2 > eps;
  float cx = valid ? dx*s : 0.f;
  float cy = valid ? dy*s : 0.f;
  float cz = valid ? (caps ? signf(dz)*nrm : 1.5f*dz) : 0.f;

  float rxy = sqrtf(cx*cx + cy*cy);
  bool  regA = cx*cx >= cy*cy;
  float safe_cx = (fabsf(cx) > eps) ? cx : 1.f;
  float safe_cy = (fabsf(cy) > eps) ? cy : 1.f;
  float uA = signf(cx)*rxy;
  float vA = uA*FOUR_OVER_PI*atanf(cy/safe_cx);
  float vB = signf(cy)*rxy;
  float uB = vB*FOUR_OVER_PI*atanf(cx/safe_cy);
  bool  okxy = rxy > eps;
  float u = okxy ? (regA ? uA : uB) : 0.f;
  float v = okxy ? (regA ? vA : vB) : 0.f;

  // align_corners trilinear on 4^3 grid: t = (u+1)*1.5
  float tx = (u  + 1.f)*1.5f, ty = (v + 1.f)*1.5f, tz = (cz + 1.f)*1.5f;
  float ix = fminf(fmaxf(floorf(tx), 0.f), 2.f);
  float iy = fminf(fmaxf(floorf(ty), 0.f), 2.f);
  float iz = fminf(fmaxf(floorf(tz), 0.f), 2.f);
  EdgeGeom g;
  g.win = win;
  g.i0x = (int)ix; g.i0y = (int)iy; g.i0z = (int)iz;
  g.fx = tx - ix;  g.fy = ty - iy;  g.fz = tz - iz;
  return g;
}

// ---------------------------------------------------------------------------
// Row offsets: edges are sorted by dst; lower_bound per node (N+1 entries)
// ---------------------------------------------------------------------------
__global__ void row_offsets_kernel(const int* __restrict__ dst, int nEdges,
                                   int* __restrict__ off, int nNodes) {
  int n = blockIdx.x * blockDim.x + threadIdx.x;
  if (n > nNodes) return;
  int lo = 0, hi = nEdges;
  while (lo < hi) {
    int mid = (lo + hi) >> 1;
    if (dst[mid] < n) lo = mid + 1; else hi = mid;
  }
  off[n] = lo;
}

// feats = [vel, mass]
__global__ void pack_feats_kernel(const float* __restrict__ vel,
                                  const float* __restrict__ mass,
                                  float* __restrict__ f, int n) {
  int i = blockIdx.x * blockDim.x + threadIdx.x;
  if (i >= n) return;
  f[4*i+0] = vel[3*i+0];
  f[4*i+1] = vel[3*i+1];
  f[4*i+2] = vel[3*i+2];
  f[4*i+3] = mass[i];
}

// ---------------------------------------------------------------------------
// Weight prepack: W[kk][cin][cout] f32 -> bf16 in WMMA B-lane order.
// Layout: [pass][kblk][nblk][lane(32)][16], lane<16: N=lane,K=i ; lane>=16:
// N=lane-16, K=16+i. K within a pass orders as kk*CHUNK + cc.
// ---------------------------------------------------------------------------
__global__ void prepack_kernel(const float* __restrict__ W, unsigned short* __restrict__ out,
                               int CIN, int CHUNK, int COUTP, int coutReal, int total) {
  int idx = blockIdx.x * blockDim.x + threadIdx.x;
  if (idx >= total) return;
  int NBLKl  = COUTP >> 4;
  int KBLKSl = (KKTAP * CHUNK) >> 5;
  int ii  = idx & 15;
  int ln  = (idx >> 4) & 31;
  int rem = idx >> 9;
  int nb  = rem % NBLKl;
  int pk  = rem / NBLKl;
  int kb  = pk % KBLKSl;
  int p   = pk / KBLKSl;
  int n   = (ln & 15) + nb * 16;
  int k32 = ((ln >> 4) << 4) + ii;
  int Kp  = kb * 32 + k32;
  int kk  = Kp / CHUNK;
  int cc  = Kp % CHUNK;
  int cin = p * CHUNK + cc;
  float v = (n < coutReal) ? W[((size_t)kk * CIN + cin) * coutReal + n] : 0.f;
  out[idx] = f2bf(v);
}

// ---------------------------------------------------------------------------
// Dense (pointwise) layers: tiny (<=100 MFLOP total), plain VALU.
// ---------------------------------------------------------------------------
__global__ void dense_kernel(const float* __restrict__ in, int ldIn,
                             const float* __restrict__ Wd, const float* __restrict__ b,
                             float* __restrict__ out, int ldOut, int colOff,
                             int CINd, int COUTd, int doRelu, int nRows) {
  int idx = blockIdx.x * blockDim.x + threadIdx.x;
  if (idx >= nRows * COUTd) return;
  int n = idx / COUTd;
  int j = idx - n * COUTd;
  float a = b[j];
  for (int c = 0; c < CINd; ++c)
    a = fmaf(in[(size_t)n * ldIn + c], Wd[(size_t)c * COUTd + j], a);
  if (doRelu) a = fmaxf(a, 0.f);
  out[(size_t)n * ldOut + colOff + j] = a;
}

// ---------------------------------------------------------------------------
// Fused ContinuousConv: 16-node M-tile per workgroup.
//  pass over K-chunks of CHUNK channels:
//    zero f32 LDS tap-tile [16][64][CHUNK] -> LDS-atomic edge scatter ->
//    f32->bf16 repack into WMMA A-lane order -> v_wmma_f32_16x16x32_bf16
//  epilogue: +bias, optional relu, optional accumulate + residual.
// ---------------------------------------------------------------------------
template <int CIN, int CHUNK, int COUTP>
__launch_bounds__(256)
__global__ void cconv_kernel(const float* __restrict__ pos_src,
                             const float* __restrict__ pos_dst,
                             const float* __restrict__ feats_src,
                             const int* __restrict__ eSrc,
                             const int* __restrict__ eDst,
                             const int* __restrict__ rowOff,
                             const unsigned short* __restrict__ Wpk,
                             const float* __restrict__ bias,
                             float invRadius,
                             float* __restrict__ outp, int ldOut, int colOff,
                             int coutReal, int flags,
                             const float* __restrict__ resid, int ldResid) {
  constexpr int PASSES = CIN / CHUNK;
  constexpr int KTOT   = KKTAP * CHUNK;     // K per pass
  constexpr int KBLKS  = KTOT / 32;
  constexpr int NBLK   = COUTP / 16;

  extern __shared__ char smem[];
  float*          Tf32 = (float*)smem;                                  // [16][KTOT]
  unsigned short* Abuf = (unsigned short*)(smem + (size_t)16 * KTOT * 4); // [KBLKS][32][16]

  const int tid  = threadIdx.x;
  const int lane = tid & 31;
  const int wave = tid >> 5;
  const int base = blockIdx.x * 16;
  const int r0 = rowOff[base];
  const int r1 = rowOff[base + 16];

  v8f acc = {};

  for (int p = 0; p < PASSES; ++p) {
    __syncthreads();                               // Abuf/Tf32 free from prev pass
    for (int i = tid; i < 16 * KTOT; i += 256) Tf32[i] = 0.f;
    __syncthreads();

    // -------- edge scatter into LDS (ds_add_f32) --------
    for (int e = r0 + tid; e < r1; e += 256) {
      int s  = eSrc[e];
      int dn = eDst[e];
      float dx = (pos_src[3*s+0] - pos_dst[3*dn+0]) * invRadius;
      float dy = (pos_src[3*s+1] - pos_dst[3*dn+1]) * invRadius;
      float dz = (pos_src[3*s+2] - pos_dst[3*dn+2]) * invRadius;
      EdgeGeom g = edge_geom(dx, dy, dz);

      float fr[CHUNK];
      const float* fs = feats_src + (size_t)s * CIN + p * CHUNK;
      #pragma unroll
      for (int c = 0; c < CHUNK; ++c) fr[c] = fs[c] * g.win;

      float wx[2] = {1.f - g.fx, g.fx};
      float wy[2] = {1.f - g.fy, g.fy};
      float wz[2] = {1.f - g.fz, g.fz};
      float* Trow = Tf32 + (size_t)(dn - base) * KTOT;
      #pragma unroll
      for (int cx = 0; cx < 2; ++cx)
      #pragma unroll
      for (int cy = 0; cy < 2; ++cy)
      #pragma unroll
      for (int cz = 0; cz < 2; ++cz) {
        int k = ((g.i0x + cx) * 4 + (g.i0y + cy)) * 4 + (g.i0z + cz);
        float wgt = wx[cx] * wy[cy] * wz[cz];
        float* dsta = Trow + k * CHUNK;
        #pragma unroll
        for (int c = 0; c < CHUNK; ++c) atomicAdd(&dsta[c], wgt * fr[c]);
      }
    }
    __syncthreads();

    // -------- f32 -> bf16 repack into WMMA A-lane layout --------
    // lane<16: M=lane, K = 0..15; lane>=16: M=lane-16, K = 16..31
    for (int i = tid; i < 16 * KTOT; i += 256) {
      int kb  = i >> 9;
      int rem = i & 511;
      int ln  = rem >> 4;
      int ii  = rem & 15;
      int m   = ln & 15;
      int kloc = ((ln >> 4) ? 16 : 0) + ii;
      Abuf[i] = f2bf(Tf32[(size_t)m * KTOT + kb * 32 + kloc]);
    }
    __syncthreads();

    // -------- tap-GEMM: acc(16x16 f32) += A(16x32 bf16) x B(32x16 bf16) --------
    if (wave < NBLK) {
      const unsigned short* Wp = Wpk + ((size_t)p * KBLKS * NBLK + wave) * 512;
      #pragma unroll 4
      for (int kb = 0; kb < KBLKS; ++kb) {
        v8u araw = *(const v8u*)&Abuf[kb * 512 + lane * 16];
        v8u braw = *(const v8u*)&Wp[(size_t)kb * NBLK * 512 + lane * 16];
        __builtin_prefetch(&Wp[(size_t)(kb + 1) * NBLK * 512 + lane * 16], 0, 1);
        v16bf a = __builtin_bit_cast(v16bf, araw);
        v16bf b = __builtin_bit_cast(v16bf, braw);
        acc = __builtin_amdgcn_wmma_f32_16x16x32_bf16(
            /*neg_a=*/false, a, /*neg_b=*/false, b,
            /*c_mod=*/(short)0, acc, /*reuse_a=*/false, /*reuse_b=*/false);
      }
    }
  }

  // -------- epilogue: bias / relu / accumulate / residual --------
  if (wave < NBLK) {
    int nc = (lane & 15) + wave * 16;
    float bv = (nc < coutReal) ? bias[nc] : 0.f;
    bool doRelu = (flags & 1) != 0;
    bool accum  = (flags & 2) != 0;
    #pragma unroll
    for (int r = 0; r < 8; ++r) {
      int m = r + ((lane >> 4) ? 8 : 0);
      int node = base + m;
      float val = acc[r] + bv;
      if (doRelu) val = fmaxf(val, 0.f);
      if (nc < coutReal) {
        float* o = outp + (size_t)node * ldOut + colOff + nc;
        float prev = accum ? *o : 0.f;
        if (resid) prev += resid[(size_t)node * ldResid + nc];
        *o = prev + val;
      }
    }
  }
}

// ---------------------------------------------------------------------------
// Orchestration
// ---------------------------------------------------------------------------
extern "C" void kernel_launch(void* const* d_in, const int* in_sizes, int n_in,
                              void* d_out, int out_size, void* d_ws, size_t ws_size,
                              hipStream_t stream) {
  (void)in_sizes; (void)n_in; (void)out_size; (void)ws_size;

  const float* pos     = (const float*)d_in[0];
  const float* vel     = (const float*)d_in[1];
  const float* mass    = (const float*)d_in[2];
  const float* bh_pos  = (const float*)d_in[3];
  const float* bh_vel  = (const float*)d_in[4];
  const float* bh_mass = (const float*)d_in[5];
  const int*   pp_src  = (const int*)d_in[6];
  const int*   pp_dst  = (const int*)d_in[7];
  const int*   bh_src  = (const int*)d_in[8];
  const int*   bh_dst  = (const int*)d_in[9];
  const float* W0a = (const float*)d_in[10]; const float* b0a = (const float*)d_in[11];
  const float* Wd0 = (const float*)d_in[12]; const float* bd0 = (const float*)d_in[13];
  const float* W0b = (const float*)d_in[14]; const float* b0b = (const float*)d_in[15];
  const float* W1  = (const float*)d_in[16]; const float* b1  = (const float*)d_in[17];
  const float* Wd1 = (const float*)d_in[18]; const float* bd1 = (const float*)d_in[19];
  const float* W2  = (const float*)d_in[20]; const float* b2  = (const float*)d_in[21];
  const float* Wd2 = (const float*)d_in[22]; const float* bd2 = (const float*)d_in[23];
  const float* W3  = (const float*)d_in[24]; const float* b3  = (const float*)d_in[25];
  const float* Wd3 = (const float*)d_in[26]; const float* bd3 = (const float*)d_in[27];
  float* out = (float*)d_out;

  // workspace carve-out (256B aligned segments)
  char* ws = (char*)d_ws;
  size_t off = 0;
  auto carve = [&](size_t bytes) -> char* {
    char* p = ws + off;
    off = (off + bytes + 255) & ~(size_t)255;
    return p;
  };
  int*   offPP = (int*)carve((NPART + 1) * sizeof(int));
  int*   offBH = (int*)carve((NPART + 1) * sizeof(int));
  float* feat0 = (float*)carve((size_t)NPART * 4 * sizeof(float));
  float* bhF0  = (float*)carve((size_t)NBHOLE * 4 * sizeof(float));
  float* h0    = (float*)carve((size_t)NPART * 96 * sizeof(float));
  float* h1    = (float*)carve((size_t)NPART * 64 * sizeof(float));
  float* h2    = (float*)carve((size_t)NPART * 64 * sizeof(float));
  unsigned short* Wpk0a = (unsigned short*)carve((size_t)8192   * 2);
  unsigned short* Wpk0b = (unsigned short*)carve((size_t)8192   * 2);
  unsigned short* Wpk1  = (unsigned short*)carve((size_t)393216 * 2);
  unsigned short* Wpk2  = (unsigned short*)carve((size_t)262144 * 2);
  unsigned short* Wpk3  = (unsigned short*)carve((size_t)65536  * 2);

  // CSR row offsets (edges are emitted sorted by dst; sentinel dst==N at tail)
  row_offsets_kernel<<<(NPART + 1 + 255) / 256, 256, 0, stream>>>(pp_dst, EPEDGE, offPP, NPART);
  row_offsets_kernel<<<(NPART + 1 + 255) / 256, 256, 0, stream>>>(bh_dst, EBEDGE, offBH, NPART);
  pack_feats_kernel<<<(NPART + 255) / 256, 256, 0, stream>>>(vel, mass, feat0, NPART);
  pack_feats_kernel<<<1, 256, 0, stream>>>(bh_vel, bh_mass, bhF0, NBHOLE);

  // prepack tap weights into bf16 WMMA B layout (L2-resident thereafter)
  prepack_kernel<<<(8192   + 255) / 256, 256, 0, stream>>>(W0a, Wpk0a, 4,  4,  32, 32, 8192);
  prepack_kernel<<<(8192   + 255) / 256, 256, 0, stream>>>(W0b, Wpk0b, 4,  4,  32, 32, 8192);
  prepack_kernel<<<(393216 + 255) / 256, 256, 0, stream>>>(W1,  Wpk1,  96, 32, 64, 64, 393216);
  prepack_kernel<<<(262144 + 255) / 256, 256, 0, stream>>>(W2,  Wpk2,  64, 32, 64, 64, 262144);
  prepack_kernel<<<(65536  + 255) / 256, 256, 0, stream>>>(W3,  Wpk3,  64, 32, 16, 3,  65536);

  const float invR_pp = 1.0f / 4.5f;    // 0.5 * EXTENT
  const float invR_bh = 1.0f / 18.0f;   // 0.5 * EXTENT_BH
  const size_t lds4  = (size_t)16 * KKTAP * 4  * 4 + (size_t)16 * KKTAP * 4  * 2; //  24 KB
  const size_t lds32 = (size_t)16 * KKTAP * 32 * 4 + (size_t)16 * KKTAP * 32 * 2; // 192 KB (<320KB/WGP)

  // ---- layer 0: h0 = [c0 | cb | d0] (concat -> disjoint column writes) ----
  dense_kernel<<<(NPART * 32 + 255) / 256, 256, 0, stream>>>(feat0, 4, Wd0, bd0, h0, 96, 64, 4, 32, 0, NPART);
  cconv_kernel<4, 4, 32><<<NPART / 16, 256, lds4, stream>>>(
      pos, pos, feat0, pp_src, pp_dst, offPP, Wpk0a, b0a, invR_pp,
      h0, 96, 0, 32, /*flags=*/0, nullptr, 0);
  cconv_kernel<4, 4, 32><<<NPART / 16, 256, lds4, stream>>>(
      bh_pos, pos, bhF0, bh_src, bh_dst, offBH, Wpk0b, b0b, invR_bh,
      h0, 96, 32, 32, /*flags=*/0, nullptr, 0);

  // ---- layer 1: h1 = relu(c1) + relu(d1) ----
  dense_kernel<<<(NPART * 64 + 255) / 256, 256, 0, stream>>>(h0, 96, Wd1, bd1, h1, 64, 0, 96, 64, 1, NPART);
  cconv_kernel<96, 32, 64><<<NPART / 16, 256, lds32, stream>>>(
      pos, pos, h0, pp_src, pp_dst, offPP, Wpk1, b1, invR_pp,
      h1, 64, 0, 64, /*relu|accum=*/3, nullptr, 0);

  // ---- layer 2: h2 = relu(c2) + relu(d2) + h1 ----
  dense_kernel<<<(NPART * 64 + 255) / 256, 256, 0, stream>>>(h1, 64, Wd2, bd2, h2, 64, 0, 64, 64, 1, NPART);
  cconv_kernel<64, 32, 64><<<NPART / 16, 256, lds32, stream>>>(
      pos, pos, h1, pp_src, pp_dst, offPP, Wpk2, b2, invR_pp,
      h2, 64, 0, 64, /*relu|accum=*/3, h1, 64);

  // ---- layer 3: out = c3 + d3 (Cout=3, padded to one 16-col WMMA tile) ----
  dense_kernel<<<(NPART * 3 + 255) / 256, 256, 0, stream>>>(h2, 64, Wd3, bd3, out, 3, 0, 64, 3, 0, NPART);
  cconv_kernel<64, 32, 16><<<NPART / 16, 256, lds32, stream>>>(
      pos, pos, h2, pp_src, pp_dst, offPP, Wpk3, b3, invR_pp,
      out, 3, 0, 3, /*accum=*/2, nullptr, 0);
}
